// MistralMoDExAttnDecoderLayer_27702539059919
// MI455X (gfx1250) — matrix operand order repaired and checked by reading, not compiled, not run on hardware
//
#include <hip/hip_runtime.h>
#include <hip/hip_bf16.h>

typedef __bf16 bf16_t;
typedef bf16_t v16bf __attribute__((ext_vector_type(16)));
typedef bf16_t v8bf  __attribute__((ext_vector_type(8)));
typedef float  v8f   __attribute__((ext_vector_type(8)));

constexpr int B_  = 2;
constexpr int S_  = 2048;
constexpr int D_  = 2048;
constexpr int H_  = 16;
constexpr int KV_ = 4;
constexpr int HD_ = 128;
constexpr int FF_ = 7168;
constexpr int M_  = B_ * S_;          // 4096 tokens

// GEMM blocking: 256 threads = 8 waves (2M x 4N), wave tile 64M x 32N.
constexpr int BM  = 128;
constexpr int BN  = 128;
constexpr int BK  = 64;
constexpr int LDK = BK + 8;           // padded LDS stride (conflict-free frag reads)

// Try the CDNA5 async global->LDS path (ASYNCcnt); set to 0 to fall back to
// plain global_load + ds_store staging if the assembler rejects the mnemonic.
#define ASYNC_CP 1

// ---------------- WMMA fragment helpers (contiguous k order, same in A and B) ----

// A-matrix 16x32 bf16: lane (l&15)=row M; contiguous 16 K-values at (l>>4)*16.
__device__ __forceinline__ v16bf ld_a_frag(const bf16_t* p) {
  v8bf lo = *(const v8bf*)(p);
  v8bf hi = *(const v8bf*)(p + 8);
  return __builtin_shufflevector(lo, hi, 0,1,2,3,4,5,6,7, 8,9,10,11,12,13,14,15);
}

// B-matrix 32x16 bf16: lane (l&15)=col N; contiguous 16 K-values at (l>>4)*16.
__device__ __forceinline__ v16bf ld_b_frag(const bf16_t* p) {
  v8bf lo = *(const v8bf*)(p);
  v8bf hi = *(const v8bf*)(p + 8);
  return __builtin_shufflevector(lo, hi, 0,1,2,3,4,5,6,7, 8,9,10,11,12,13,14,15);
}

__device__ __forceinline__ v8f wmma_bf16(v16bf a, v16bf b, v8f c) {
  return __builtin_amdgcn_wmma_f32_16x16x32_bf16(false, a, false, b,
                                                 (short)0, c, false, false);
}

// ---------------- global -> LDS staging primitives ---------------------------

__device__ __forceinline__ void cp16_g2l(bf16_t* ldst, const bf16_t* gsrc) {
#if ASYNC_CP
  unsigned loff = (unsigned)(size_t)ldst;      // LDS byte offset (low 32 bits)
  asm volatile("global_load_async_to_lds_b128 %0, %1, off"
               :: "v"(loff), "v"(gsrc) : "memory");
#else
  *(v8bf*)ldst = *(const v8bf*)gsrc;
#endif
}

__device__ __forceinline__ void cp_commit() {
#if ASYNC_CP
  asm volatile("s_wait_asynccnt 0" ::: "memory");
#endif
  __syncthreads();
}

// ---------------- LDS-staged, double-buffered GEMM core ----------------------
// Accumulates c[4][2] = 64M x 32N wave tile of A[M,K] @ W[N,K]^T.
__device__ __forceinline__ void gemm_core_staged(const bf16_t* __restrict__ A,
                                                 const bf16_t* __restrict__ W,
                                                 int K, v8f (&c)[4][2]) {
  __shared__ alignas(16) bf16_t As[2][BM][LDK];
  __shared__ alignas(16) bf16_t Bs[2][BM][LDK];
  const int tid  = threadIdx.x;
  const int wave = tid >> 5, lane = tid & 31;
  const int colc = lane & 15, half = lane >> 4;
  const int waveM = (wave >> 2) * 64;
  const int waveN = (wave & 3) * 32;
  const size_t mBase = (size_t)blockIdx.y * BM;
  const size_t nBase = (size_t)blockIdx.x * BN;

  const int srow = tid >> 3;          // 0..31, 4 iterations cover 128 rows
  const int skc  = (tid & 7) * 8;     // 16B chunk within K-slab

  auto stage = [&](int buf, int k0) {
    #pragma unroll
    for (int i = 0; i < 4; i++) {
      int r = srow + i * 32;
      cp16_g2l(&As[buf][r][skc], A + (mBase + r) * K + k0 + skc);
      cp16_g2l(&Bs[buf][r][skc], W + (nBase + r) * K + k0 + skc);
    }
  };

  stage(0, 0);
  cp_commit();
  const int nk = K / BK;
  for (int ks = 0; ks < nk; ks++) {
    int buf = ks & 1;
    if (ks + 1 < nk) stage(buf ^ 1, (ks + 1) * BK);
    const bf16_t* aw = &As[buf][waveM][0];
    const bf16_t* bw = &Bs[buf][waveN][0];
    #pragma unroll
    for (int kk = 0; kk < BK; kk += 32) {
      v16bf fa[4], fb[2];
      #pragma unroll
      for (int mi = 0; mi < 4; mi++)
        fa[mi] = ld_a_frag(aw + (size_t)(mi * 16 + colc) * LDK + kk + half * 16);
      #pragma unroll
      for (int ni = 0; ni < 2; ni++)
        fb[ni] = ld_b_frag(bw + (size_t)(ni * 16 + colc) * LDK + kk + half * 16);
      #pragma unroll
      for (int mi = 0; mi < 4; mi++)
        #pragma unroll
        for (int ni = 0; ni < 2; ni++)
          c[mi][ni] = wmma_bf16(fa[mi], fb[ni], c[mi][ni]);
    }
    cp_commit();
  }
}

// ---------------- elementwise / norm kernels ---------------------------------

__global__ void __launch_bounds__(256) cvt_f32_to_bf16(const float* __restrict__ x,
                                                       bf16_t* __restrict__ y, int n) {
  int i = blockIdx.x * 256 + threadIdx.x;
  if (i < n) y[i] = (bf16_t)x[i];
}

__global__ void __launch_bounds__(256) rmsnorm_bf16(const float* __restrict__ x,
                                                    const float* __restrict__ w,
                                                    bf16_t* __restrict__ y) {
  const int row = blockIdx.x;
  const float* xr = x + (size_t)row * D_;
  float ss = 0.f;
  for (int i = threadIdx.x; i < D_; i += 256) { float v = xr[i]; ss += v * v; }
  #pragma unroll
  for (int off = 16; off > 0; off >>= 1) ss += __shfl_xor(ss, off, 32);
  __shared__ float red[8];
  __shared__ float sinv;
  int wave = threadIdx.x >> 5, lane = threadIdx.x & 31;
  if (lane == 0) red[wave] = ss;
  __syncthreads();
  if (threadIdx.x == 0) {
    float t = 0.f;
    #pragma unroll
    for (int i = 0; i < 8; i++) t += red[i];
    sinv = rsqrtf(t / (float)D_ + 1e-5f);
  }
  __syncthreads();
  float inv = sinv;
  bf16_t* yr = y + (size_t)row * D_;
  for (int i = threadIdx.x; i < D_; i += 256) yr[i] = (bf16_t)(xr[i] * inv * w[i]);
}

// src: [B,S,nh,HD] f32 (GEMM output) -> dst: [B,nh,S,HD] bf16 with RoPE applied
__global__ void __launch_bounds__(256) rope_pack(const float* __restrict__ src,
                                                 bf16_t* __restrict__ dst, int nh) {
  int t = blockIdx.x * 256 + threadIdx.x;
  int total = B_ * S_ * nh * (HD_ / 2);
  if (t >= total) return;
  int i = t & 63; int rest = t >> 6;
  int h = rest % nh; rest /= nh;
  int s = rest % S_; int b = rest / S_;
  float inv = __powf(10000.0f, -(float)(2 * i) / (float)HD_);
  float ang = (float)s * inv;
  float cs = __cosf(ang), sn = __sinf(ang);
  const float* p = src + ((size_t)(b * S_ + s) * nh + h) * HD_;
  float x1 = p[i], x2 = p[i + 64];
  bf16_t* q = dst + ((size_t)(b * nh + h) * S_ + s) * HD_;
  q[i]      = (bf16_t)(x1 * cs - x2 * sn);
  q[i + 64] = (bf16_t)(x2 * cs + x1 * sn);
}

// src: [B,S,KV,HD] f32 -> dst: [B,KV,HD,S] bf16 (V transposed for B-fragments)
__global__ void __launch_bounds__(256) pack_vt(const float* __restrict__ src,
                                               bf16_t* __restrict__ dst) {
  int t = blockIdx.x * 256 + threadIdx.x;
  int total = B_ * S_ * KV_ * HD_;
  if (t >= total) return;
  int d = t % HD_; int rest = t / HD_;
  int kv = rest % KV_; rest /= KV_;
  int s = rest % S_; int b = rest / S_;
  dst[((size_t)(b * KV_ + kv) * HD_ + d) * S_ + s] = (bf16_t)src[t];
}

// ---------------- GEMM kernels (block tile 128x128, 8 waves) ------------------

__global__ void __launch_bounds__(256) gemm_f32out(const bf16_t* __restrict__ A,
                                                   const bf16_t* __restrict__ W,
                                                   float* __restrict__ C, int K, int N) {
  int lane = threadIdx.x & 31, wave = threadIdx.x >> 5;
  int colc = lane & 15, half = lane >> 4;
  int m0 = blockIdx.y * BM + (wave >> 2) * 64;
  int n0 = blockIdx.x * BN + (wave & 3) * 32;
  v8f c[4][2] = {};
  gemm_core_staged(A, W, K, c);
  #pragma unroll
  for (int mi = 0; mi < 4; mi++)
    #pragma unroll
    for (int ni = 0; ni < 2; ni++)
      #pragma unroll
      for (int r = 0; r < 8; r++)
        C[(size_t)(m0 + mi * 16 + half * 8 + r) * N + (n0 + ni * 16 + colc)] = c[mi][ni][r];
}

__global__ void __launch_bounds__(256) gemm_oproj(const bf16_t* __restrict__ A,
                                                  const bf16_t* __restrict__ W,
                                                  const float* __restrict__ resid,
                                                  float* __restrict__ out, int K, int N) {
  int lane = threadIdx.x & 31, wave = threadIdx.x >> 5;
  int colc = lane & 15, half = lane >> 4;
  int m0 = blockIdx.y * BM + (wave >> 2) * 64;
  int n0 = blockIdx.x * BN + (wave & 3) * 32;
  v8f c[4][2] = {};
  gemm_core_staged(A, W, K, c);
  #pragma unroll
  for (int mi = 0; mi < 4; mi++)
    #pragma unroll
    for (int ni = 0; ni < 2; ni++)
      #pragma unroll
      for (int r = 0; r < 8; r++) {
        size_t idx = (size_t)(m0 + mi * 16 + half * 8 + r) * N + (n0 + ni * 16 + colc);
        out[idx] = resid[idx] + c[mi][ni][r];
      }
}

// down-proj with MoD epilogue: out[t,n] += mask[t] ? acc * (0.5 + (score[t]-0.5)*0.7) : 0
__global__ void __launch_bounds__(256) gemm_down(const bf16_t* __restrict__ A,
                                                 const bf16_t* __restrict__ W,
                                                 const int* __restrict__ mask,
                                                 const float* __restrict__ score,
                                                 float* __restrict__ out, int K, int N) {
  int lane = threadIdx.x & 31, wave = threadIdx.x >> 5;
  int colc = lane & 15, half = lane >> 4;
  int m0 = blockIdx.y * BM + (wave >> 2) * 64;
  int n0 = blockIdx.x * BN + (wave & 3) * 32;
  v8f c[4][2] = {};
  gemm_core_staged(A, W, K, c);
  #pragma unroll
  for (int mi = 0; mi < 4; mi++)
    #pragma unroll
    for (int r = 0; r < 8; r++) {
      int t = m0 + mi * 16 + half * 8 + r;
      if (mask[t]) {
        float sc = 0.5f + (score[t] - 0.5f) * 0.7f;
        #pragma unroll
        for (int ni = 0; ni < 2; ni++)
          out[(size_t)t * N + (n0 + ni * 16 + colc)] += c[mi][ni][r] * sc;
      }
    }
}

// fused gate/up: G = silu(x@Wg^T) * (x@Wu^T) -> bf16 [M,FF], A/Wg/Wu tiles double-buffered in LDS
__global__ void __launch_bounds__(256) gemm_gateup(const bf16_t* __restrict__ A,
                                                   const bf16_t* __restrict__ Wg,
                                                   const bf16_t* __restrict__ Wu,
                                                   bf16_t* __restrict__ G, int K, int N) {
  __shared__ alignas(16) bf16_t As[2][BM][LDK];
  __shared__ alignas(16) bf16_t Gs[2][BM][LDK];
  __shared__ alignas(16) bf16_t Us[2][BM][LDK];
  const int tid  = threadIdx.x;
  const int wave = tid >> 5, lane = tid & 31;
  const int colc = lane & 15, half = lane >> 4;
  const int waveM = (wave >> 2) * 64;
  const int waveN = (wave & 3) * 32;
  const size_t mBase = (size_t)blockIdx.y * BM;
  const size_t nBase = (size_t)blockIdx.x * BN;
  const int srow = tid >> 3;
  const int skc  = (tid & 7) * 8;

  auto stage = [&](int buf, int k0) {
    #pragma unroll
    for (int i = 0; i < 4; i++) {
      int r = srow + i * 32;
      cp16_g2l(&As[buf][r][skc], A  + (mBase + r) * K + k0 + skc);
      cp16_g2l(&Gs[buf][r][skc], Wg + (nBase + r) * K + k0 + skc);
      cp16_g2l(&Us[buf][r][skc], Wu + (nBase + r) * K + k0 + skc);
    }
  };

  v8f cg[4][2] = {}, cu[4][2] = {};
  stage(0, 0);
  cp_commit();
  const int nk = K / BK;
  for (int ks = 0; ks < nk; ks++) {
    int buf = ks & 1;
    if (ks + 1 < nk) stage(buf ^ 1, (ks + 1) * BK);
    const bf16_t* aw = &As[buf][waveM][0];
    const bf16_t* gw = &Gs[buf][waveN][0];
    const bf16_t* uw = &Us[buf][waveN][0];
    #pragma unroll
    for (int kk = 0; kk < BK; kk += 32) {
      v16bf fa[4], fg[2], fu[2];
      #pragma unroll
      for (int mi = 0; mi < 4; mi++)
        fa[mi] = ld_a_frag(aw + (size_t)(mi * 16 + colc) * LDK + kk + half * 16);
      #pragma unroll
      for (int ni = 0; ni < 2; ni++) {
        fg[ni] = ld_b_frag(gw + (size_t)(ni * 16 + colc) * LDK + kk + half * 16);
        fu[ni] = ld_b_frag(uw + (size_t)(ni * 16 + colc) * LDK + kk + half * 16);
      }
      #pragma unroll
      for (int mi = 0; mi < 4; mi++)
        #pragma unroll
        for (int ni = 0; ni < 2; ni++) {
          cg[mi][ni] = wmma_bf16(fa[mi], fg[ni], cg[mi][ni]);
          cu[mi][ni] = wmma_bf16(fa[mi], fu[ni], cu[mi][ni]);
        }
    }
    cp_commit();
  }

  const int m0 = (int)mBase + waveM;
  const int n0 = (int)nBase + waveN;
  #pragma unroll
  for (int mi = 0; mi < 4; mi++)
    #pragma unroll
    for (int ni = 0; ni < 2; ni++)
      #pragma unroll
      for (int r = 0; r < 8; r++) {
        float gv = cg[mi][ni][r];
        float uv = cu[mi][ni][r];
        float sg = gv / (1.f + __expf(-gv));
        G[(size_t)(m0 + mi * 16 + half * 8 + r) * N + (n0 + ni * 16 + colc)]
            = (bf16_t)(sg * uv);
      }
}

// ---------------- flash attention (per-wave 16 query rows) --------------------
// Q: [B,H,S,HD] bf16 (roped), K: [B,KV,S,HD] bf16 (roped), Vt: [B,KV,HD,S] bf16
// O: [B,S,H*HD] bf16 (A-matrix layout for o-proj GEMM)
__global__ void __launch_bounds__(128) attn_fwd(const bf16_t* __restrict__ Q,
                                                const bf16_t* __restrict__ Kc,
                                                const bf16_t* __restrict__ Vt,
                                                bf16_t* __restrict__ O) {
  constexpr int QT = S_ / 64;
  int bid = blockIdx.x;
  int qt = bid % QT;
  int h  = (bid / QT) % H_;
  int b  = bid / (QT * H_);
  int kv = h / (H_ / KV_);

  int wave = threadIdx.x >> 5, lane = threadIdx.x & 31;
  int colc = lane & 15, half = lane >> 4;
  int q0 = qt * 64 + wave * 16;

  const bf16_t* qb = Q + ((size_t)(b * H_ + h) * S_ + q0 + colc) * HD_ + half * 16;
  v16bf aq[4];
  #pragma unroll
  for (int j = 0; j < 4; j++) aq[j] = ld_a_frag(qb + j * 32);

  const bf16_t* kbase = Kc + (size_t)(b * KV_ + kv) * S_ * HD_;
  const bf16_t* vbase = Vt + (size_t)(b * KV_ + kv) * HD_ * S_;

  float mrun[8], lrun[8];
  #pragma unroll
  for (int r = 0; r < 8; r++) { mrun[r] = -1e30f; lrun[r] = 0.f; }
  v8f o[8] = {};

  __shared__ bf16_t ptile[4][16][32];   // per-wave P staging (C-layout -> A-layout)

  const float sc = 0.08838834764831845f;   // 1/sqrt(128)
  int nkt = (q0 + 47) >> 5;                // causal: keys <= q0+15

  for (int kt = 0; kt < nkt; kt++) {
    v8f s0 = {}, s1 = {};
    const bf16_t* kp0 = kbase + (size_t)(kt * 32 + colc) * HD_ + half * 16;
    const bf16_t* kp1 = kp0 + (size_t)16 * HD_;
    #pragma unroll
    for (int j = 0; j < 4; j++) {
      v16bf b0 = ld_b_frag(kp0 + j * 32);
      v16bf b1 = ld_b_frag(kp1 + j * 32);
      s0 = wmma_bf16(aq[j], b0, s0);
      s1 = wmma_bf16(aq[j], b1, s1);
    }

    int key0 = kt * 32 + colc;
    int key1 = key0 + 16;
    float alpha[8];
    #pragma unroll
    for (int r = 0; r < 8; r++) {
      int qidx = q0 + half * 8 + r;
      float v0 = (key0 <= qidx) ? s0[r] * sc : -1e30f;
      float v1 = (key1 <= qidx) ? s1[r] * sc : -1e30f;
      float mx = fmaxf(v0, v1);
      mx = fmaxf(mx, __shfl_xor(mx, 1, 32));
      mx = fmaxf(mx, __shfl_xor(mx, 2, 32));
      mx = fmaxf(mx, __shfl_xor(mx, 4, 32));
      mx = fmaxf(mx, __shfl_xor(mx, 8, 32));
      float mnew = fmaxf(mrun[r], mx);
      float e0 = __expf(v0 - mnew);
      float e1 = __expf(v1 - mnew);
      float sum = e0 + e1;
      sum += __shfl_xor(sum, 1, 32);
      sum += __shfl_xor(sum, 2, 32);
      sum += __shfl_xor(sum, 4, 32);
      sum += __shfl_xor(sum, 8, 32);
      float a = __expf(mrun[r] - mnew);
      lrun[r] = lrun[r] * a + sum;
      mrun[r] = mnew;
      alpha[r] = a;
      ptile[wave][half * 8 + r][colc]      = (bf16_t)e0;
      ptile[wave][half * 8 + r][colc + 16] = (bf16_t)e1;
    }
    #pragma unroll
    for (int f = 0; f < 8; f++) {
      v8f t = o[f];
      #pragma unroll
      for (int r = 0; r < 8; r++) t[r] *= alpha[r];
      o[f] = t;
    }

    v16bf ap = ld_a_frag(&ptile[wave][colc][half * 16]);

    #pragma unroll
    for (int f = 0; f < 8; f++) {
      const bf16_t* vp = vbase + (size_t)(f * 16 + colc) * S_ + kt * 32 + half * 16;
      v16bf bv = ld_b_frag(vp);
      o[f] = wmma_bf16(ap, bv, o[f]);
    }
  }

  #pragma unroll
  for (int f = 0; f < 8; f++)
    #pragma unroll
    for (int r = 0; r < 8; r++) {
      float val = o[f][r] / lrun[r];
      O[(size_t)(b * S_ + q0 + half * 8 + r) * (H_ * HD_) + h * HD_ + f * 16 + colc]
          = (bf16_t)val;
    }
}

// ---------------- host orchestration -----------------------------------------

extern "C" void kernel_launch(void* const* d_in, const int* in_sizes, int n_in,
                              void* d_out, int out_size, void* d_ws, size_t ws_size,
                              hipStream_t stream) {
  (void)in_sizes; (void)n_in; (void)out_size; (void)ws_size;
  const float* hidden = (const float*)d_in[0];
  const int*   mask   = (const int*)  d_in[1];
  const float* score  = (const float*)d_in[2];
  const float* ln1    = (const float*)d_in[3];
  const float* ln2    = (const float*)d_in[4];
  const float* qw     = (const float*)d_in[5];
  const float* kw     = (const float*)d_in[6];
  const float* vw     = (const float*)d_in[7];
  const float* ow     = (const float*)d_in[8];
  const float* gw     = (const float*)d_in[9];
  const float* uw     = (const float*)d_in[10];
  const float* dw     = (const float*)d_in[11];
  float* out = (float*)d_out;

  char* ws = (char*)d_ws;
  size_t off = 0;
  auto alloc = [&](size_t bytes) -> char* {
    char* p = ws + off;
    off = (off + bytes + 255) & ~(size_t)255;
    return p;
  };

  bf16_t* xn  = (bf16_t*)alloc((size_t)M_ * D_ * 2);            // rmsnorm1 out
  float*  qf  = (float*) alloc((size_t)M_ * H_  * HD_ * 4);     // q pre-rope
  float*  kf  = (float*) alloc((size_t)M_ * KV_ * HD_ * 4);     // k pre-rope
  float*  vf  = (float*) alloc((size_t)M_ * KV_ * HD_ * 4);     // v pre-pack
  bf16_t* qp  = (bf16_t*)alloc((size_t)M_ * H_  * HD_ * 2);     // [B,H,S,HD]
  bf16_t* kp  = (bf16_t*)alloc((size_t)M_ * KV_ * HD_ * 2);     // [B,KV,S,HD]
  bf16_t* vtp = (bf16_t*)alloc((size_t)M_ * KV_ * HD_ * 2);     // [B,KV,HD,S]
  bf16_t* at  = (bf16_t*)alloc((size_t)M_ * H_  * HD_ * 2);     // attn out
  bf16_t* hn  = (bf16_t*)alloc((size_t)M_ * D_ * 2);            // rmsnorm2 out
  bf16_t* gb  = (bf16_t*)alloc((size_t)M_ * FF_ * 2);           // silu(g)*u
  bf16_t* wq  = (bf16_t*)alloc((size_t)H_  * HD_ * D_ * 2);
  bf16_t* wk  = (bf16_t*)alloc((size_t)KV_ * HD_ * D_ * 2);
  bf16_t* wv  = (bf16_t*)alloc((size_t)KV_ * HD_ * D_ * 2);
  bf16_t* wo  = (bf16_t*)alloc((size_t)D_ * H_ * HD_ * 2);
  bf16_t* wg  = (bf16_t*)alloc((size_t)FF_ * D_ * 2);
  bf16_t* wu  = (bf16_t*)alloc((size_t)FF_ * D_ * 2);
  bf16_t* wd  = (bf16_t*)alloc((size_t)D_ * FF_ * 2);

  auto cvt = [&](const float* s, bf16_t* d, size_t n) {
    cvt_f32_to_bf16<<<(unsigned)((n + 255) / 256), 256, 0, stream>>>(s, d, (int)n);
  };
  cvt(qw, wq, (size_t)H_  * HD_ * D_);
  cvt(kw, wk, (size_t)KV_ * HD_ * D_);
  cvt(vw, wv, (size_t)KV_ * HD_ * D_);
  cvt(ow, wo, (size_t)D_ * H_ * HD_);
  cvt(gw, wg, (size_t)FF_ * D_);
  cvt(uw, wu, (size_t)FF_ * D_);
  cvt(dw, wd, (size_t)D_ * FF_);

  // 1) xn = rmsnorm(hidden, ln1)
  rmsnorm_bf16<<<M_, 256, 0, stream>>>(hidden, ln1, xn);

  // 2) q/k/v projections
  gemm_f32out<<<dim3((H_ * HD_) / BN, M_ / BM), 256, 0, stream>>>(xn, wq, qf, D_, H_ * HD_);
  gemm_f32out<<<dim3((KV_ * HD_) / BN, M_ / BM), 256, 0, stream>>>(xn, wk, kf, D_, KV_ * HD_);
  gemm_f32out<<<dim3((KV_ * HD_) / BN, M_ / BM), 256, 0, stream>>>(xn, wv, vf, D_, KV_ * HD_);

  // 3) rope + layout packs
  rope_pack<<<(B_ * S_ * H_  * (HD_ / 2)) / 256, 256, 0, stream>>>(qf, qp, H_);
  rope_pack<<<(B_ * S_ * KV_ * (HD_ / 2)) / 256, 256, 0, stream>>>(kf, kp, KV_);
  pack_vt  <<<(B_ * S_ * KV_ * HD_) / 256, 256, 0, stream>>>(vf, vtp);

  // 4) flash attention
  attn_fwd<<<B_ * H_ * (S_ / 64), 128, 0, stream>>>(qp, kp, vtp, at);

  // 5) h = hidden + attn @ o_w^T   (writes every element of d_out)
  gemm_oproj<<<dim3(D_ / BN, M_ / BM), 256, 0, stream>>>(at, wo, hidden, out, H_ * HD_, D_);

  // 6) hn = rmsnorm(h, ln2)
  rmsnorm_bf16<<<M_, 256, 0, stream>>>(out, ln2, hn);

  // 7) g = silu(hn@gate^T) * (hn@up^T)
  gemm_gateup<<<dim3(FF_ / BN, M_ / BM), 256, 0, stream>>>(hn, wg, wu, gb, D_, FF_);

  // 8) out += MoD-masked (g @ down^T) * score_scale
  gemm_down<<<dim3(D_ / BN, M_ / BM), 256, 0, stream>>>(gb, wd, mask, score, out, FF_, D_);
}
